// QEffGrok1MultiHeadAttention_52269751992571
// MI455X (gfx1250) — matrix operand order repaired and checked by reading, not compiled, not run on hardware
//
#include <hip/hip_runtime.h>
#include <hip/hip_fp16.h>
#include <math.h>
#include <stdint.h>

// ---- problem constants (match reference) ----
#define B_    2
#define S_    2048
#define D_    4096
#define H_    32
#define KVH_  8
#define HD_   128
#define BS_   (B_ * S_)          // 4096 rows
#define QN_   (H_ * HD_)         // 4096
#define KN_   (KVH_ * HD_)       // 1024
#define MULT_ 0.08838834764831845f
#define MAXV_ 30.0f

typedef _Float16 half_t;
typedef __attribute__((ext_vector_type(16))) _Float16 v16h;
typedef __attribute__((ext_vector_type(8)))  _Float16 v8h;
typedef __attribute__((ext_vector_type(8)))  float    v8f;
typedef __attribute__((ext_vector_type(4)))  unsigned int u32x4;
typedef __attribute__((ext_vector_type(8)))  unsigned int u32x8;

union AF { v16h v; v8h h[2]; };

// ------------------------------------------------------------------
// CDNA5 async global->LDS copy (ASYNCcnt-tracked, 16B per op).
// ------------------------------------------------------------------
__device__ __forceinline__ void async_b128(unsigned lds_addr, const void* gaddr) {
    asm volatile("global_load_async_to_lds_b128 %0, %1, off"
                 :: "v"(lds_addr), "v"(gaddr) : "memory");
}
__device__ __forceinline__ unsigned lds_off(const void* p) {
    return (unsigned)(uintptr_t)p;
}

// ------------------------------------------------------------------
// CDNA5 Tensor Data Mover: 2D tile load, descriptor in SGPR groups.
// ------------------------------------------------------------------
__device__ __forceinline__ void tdm_load_2d(u32x4 g0, u32x8 g1) {
    asm volatile("tensor_load_to_lds %0, %1"
                 :: "s"(g0), "s"(g1) : "memory");
}

// ------------------------------------------------------------------
// f32 -> f16 cast (grid-stride)
// ------------------------------------------------------------------
__global__ void cast_f32_f16(const float* __restrict__ in,
                             half_t* __restrict__ out, long n) {
    long i = (long)blockIdx.x * blockDim.x + threadIdx.x;
    long stride = (long)gridDim.x * blockDim.x;
    for (; i < n; i += stride) out[i] = (half_t)in[i];
}

// ------------------------------------------------------------------
// C(MxN,f32) = A(MxK,f16,row) * B(KxN,f16,row)
// Block: 256 threads (8 waves), 128x128 tile, BK=32.
// A panel: double-buffered TDM (tensor_load_to_lds, TENSORcnt).
// B panel: reg-staged + stored transposed (BsT[n][k]) -> ds_load_b128 frags.
// Each wave: 32 rows x 64 cols = 8 x v_wmma_f32_16x16x32_f16 per k-step.
// ------------------------------------------------------------------
#define BM_ 128
#define BN_ 128
#define BK_ 32

__global__ __launch_bounds__(256)
void gemm_f16_wmma(const half_t* __restrict__ A,
                   const half_t* __restrict__ Bw,
                   float* __restrict__ C,
                   int M, int N, int K) {
    __shared__ __align__(16) half_t As [2][BM_ * BK_];   // 2 x 8KB
    __shared__ __align__(16) half_t BsT[BN_ * BK_];      // 8KB, [n][k]

    const int tid  = threadIdx.x;
    const int lane = tid & 31;
    const int wave = tid >> 5;
    const int wm   = wave & 3;            // 4 wave-rows
    const int wn   = wave >> 2;           // 2 wave-cols
    const int m0   = blockIdx.y * BM_;
    const int n0   = blockIdx.x * BN_;

    // B staging: 32 rows x 128 halves (8 threads/row, 16 halves each)
    const int brow = tid >> 3;
    const int bseg = (tid & 7) * 16;

    // WMMA fragment lane addressing (wave32 16-bit layouts)
    const int fm    = lane & 15;          // A-frag row
    const int khalf = (lane >> 4) * 8;    // A-frag K sub-offset
    const int fn    = lane & 15;          // B-frag column
    const int kbB   = (lane >> 4) * 8;    // B-frag K base
    const int mbase = (lane >> 4) * 8;    // D-tile row base

    v8f acc[2][4];
    #pragma unroll
    for (int mt = 0; mt < 2; ++mt)
        #pragma unroll
        for (int nt = 0; nt < 4; ++nt) acc[mt][nt] = (v8f){};

    // ---- TDM descriptor group1 (k-invariant): 2-byte elems, tile 32x128,
    //      tensor dims KxM, line stride K ----
    u32x8 g1;
    g1[0] = 1u << 16;                                            // data_size=2B
    g1[1] = ((unsigned)K & 0xFFFFu) << 16;                       // tensor_dim0 lo
    g1[2] = (((unsigned)K >> 16) & 0xFFFFu)
          | (((unsigned)M & 0xFFFFu) << 16);                     // dim0 hi | dim1 lo
    g1[3] = (((unsigned)M >> 16) & 0xFFFFu) | ((unsigned)BK_ << 16); // dim1 hi | tile_dim0
    g1[4] = (unsigned)BM_;                                       // tile_dim1 (tile_dim2=0)
    g1[5] = (unsigned)K;                                         // dim0_stride lo32
    g1[6] = 0u;
    g1[7] = 0u;

    auto stageA_tdm = [&](int buf, int k) {
        unsigned long long ga =
            (unsigned long long)(uintptr_t)(A + (long)m0 * K + k);
        u32x4 g0;
        g0[0] = 1u;                                   // count=1, user mode
        g0[1] = lds_off(&As[buf][0]);                 // lds_addr
        g0[2] = (unsigned)(ga & 0xFFFFFFFFu);         // global_addr[31:0]
        g0[3] = (unsigned)((ga >> 32) & 0x01FFFFFFu)  // global_addr[56:32]
              | 0x80000000u;                          // type=2 ("image")
        tdm_load_2d(g0, g1);
    };

    if (wave == 0) stageA_tdm(0, 0);

    for (int k = 0; k < K; k += BK_) {
        const int buf   = (k / BK_) & 1;
        const bool more = (k + BK_) < K;
        if (more && wave == 0) stageA_tdm(buf ^ 1, k + BK_);  // DMA next panel

        // B panel (current k): two contiguous 128-bit global loads into regs
        const half_t* gb = Bw + (long)(k + brow) * N + n0 + bseg;
        v8h rb0 = *reinterpret_cast<const v8h*>(gb);
        v8h rb1 = *reinterpret_cast<const v8h*>(gb + 8);
        if (more) __builtin_prefetch(gb + (long)BK_ * N, 0, 1);  // next B panel

        // current A panel landed (next panel's TDM may stay in flight)
        if (wave == 0) {
            if (more) __builtin_amdgcn_s_wait_tensorcnt(1);
            else      __builtin_amdgcn_s_wait_tensorcnt(0);
        }
        __syncthreads();   // A(cur) visible everywhere; BsT readers done

        // scatter-transpose B into BsT[n][k]
        #pragma unroll
        for (int j = 0; j < 8; ++j) BsT[(bseg + j)     * BK_ + brow] = rb0[j];
        #pragma unroll
        for (int j = 0; j < 8; ++j) BsT[(bseg + 8 + j) * BK_ + brow] = rb1[j];
        __syncthreads();   // BsT(cur) visible

        // A fragments: two ds_load_b128 per 16-row tile
        AF afr[2];
        #pragma unroll
        for (int mt = 0; mt < 2; ++mt) {
            const half_t* ar = &As[buf][(wm * 32 + mt * 16 + fm) * BK_];
            afr[mt].h[0] = *reinterpret_cast<const v8h*>(ar + khalf);
            afr[mt].h[1] = *reinterpret_cast<const v8h*>(ar + 16 + khalf);
        }
        // B fragments: two ds_load_b128 per 16-col tile (contiguous K)
        #pragma unroll
        for (int nt = 0; nt < 4; ++nt) {
            AF bf;
            const half_t* bp = &BsT[(wn * 64 + nt * 16 + fn) * BK_ + kbB];
            bf.h[0] = *reinterpret_cast<const v8h*>(bp);
            bf.h[1] = *reinterpret_cast<const v8h*>(bp + 16);
            #pragma unroll
            for (int mt = 0; mt < 2; ++mt)
                acc[mt][nt] = __builtin_amdgcn_wmma_f32_16x16x32_f16(
                    false, afr[mt].v, false, bf.v, (short)0, acc[mt][nt],
                    false, false);
        }
        __syncthreads();   // panel reads done before As[buf] is DMA-overwritten
    }

    // D tile stores (f32 16x16 D layout)
    #pragma unroll
    for (int mt = 0; mt < 2; ++mt) {
        #pragma unroll
        for (int nt = 0; nt < 4; ++nt) {
            const int cm = m0 + wm * 32 + mt * 16 + mbase;
            const int cn = n0 + wn * 64 + nt * 16 + fn;
            #pragma unroll
            for (int r = 0; r < 8; ++r)
                C[(long)(cm + r) * N + cn] = acc[mt][nt][r];
        }
    }
}

// ------------------------------------------------------------------
// RoPE + relayout: src f32 (B,S,heads,HD) -> dst f16 (B,heads,S,HD)
// ------------------------------------------------------------------
__global__ void rope_relayout(const float* __restrict__ src,
                              half_t* __restrict__ dst, int heads) {
    long total = (long)BS_ * heads * 64;
    long i = (long)blockIdx.x * blockDim.x + threadIdx.x;
    long stride = (long)gridDim.x * blockDim.x;
    for (; i < total; i += stride) {
        int  d = (int)(i & 63);
        long t = i >> 6;
        int  h = (int)(t % heads); t /= heads;
        int  s = (int)(t % S_);
        int  b = (int)(t / S_);

        float inv = __powf(10000.0f, -((float)(2 * d) / (float)HD_));
        float fr  = (float)s * inv;
        float cs  = __cosf(fr);
        float sn  = __sinf(fr);

        const float* p = src + (((long)(b * S_ + s) * heads + h) * HD_);
        float x0 = p[d];
        float x1 = p[d + 64];

        half_t* q = dst + ((((long)b * heads + h) * S_ + s) * HD_);
        q[d]      = (half_t)(x0 * cs - x1 * sn);
        q[d + 64] = (half_t)(x1 * cs + x0 * sn);
    }
}

// ------------------------------------------------------------------
// V cast + transpose: f32 (B,S,KVH,HD) -> f16 VT (B,KVH,HD,S)
// ------------------------------------------------------------------
__global__ void cast_transpose_v(const float* __restrict__ src,
                                 half_t* __restrict__ dst) {
    long total = (long)BS_ * KVH_ * HD_;
    long i = (long)blockIdx.x * blockDim.x + threadIdx.x;
    long stride = (long)gridDim.x * blockDim.x;
    for (; i < total; i += stride) {
        int  d  = (int)(i % HD_);
        long t  = i / HD_;
        int  kv = (int)(t % KVH_); t /= KVH_;
        int  s  = (int)(t % S_);
        int  b  = (int)(t / S_);
        dst[(((long)b * KVH_ + kv) * HD_ + d) * S_ + s] =
            (half_t)src[(((long)(b * S_ + s)) * KVH_ + kv) * HD_ + d];
    }
}

// ------------------------------------------------------------------
// Flash-style attention with tanh soft-cap + causal mask.
// Q: (B,H,S,HD) f16, K: (B,KVH,S,HD) f16, VT: (B,KVH,HD,S) f16,
// O: (B,S,H*HD) f16.  One wave = one 16-query tile; key steps of 32.
// Q tile staged once to LDS via async DMA, fragments hoisted to regs.
// ------------------------------------------------------------------
__global__ __launch_bounds__(256)
void attn_wmma(const half_t* __restrict__ Q,
               const half_t* __restrict__ Kt,
               const half_t* __restrict__ VT,
               half_t* __restrict__ O) {
    __shared__ __align__(16) half_t Qs[8][16 * HD_];  // 8 x 4KB per-wave Q tiles
    __shared__ __align__(16) half_t Pt[8][16][32];    // per-wave P staging

    const int lane = threadIdx.x & 31;
    const int wave = threadIdx.x >> 5;
    const int b    = blockIdx.z;
    const int h    = blockIdx.y;
    const int kvh  = h / (H_ / KVH_);
    const int q0   = (blockIdx.x * 8 + wave) * 16;

    const half_t* Qb = Q  + (((long)b * H_   + h)   * S_) * HD_;
    const half_t* Kb = Kt + (((long)b * KVH_ + kvh) * S_) * HD_;
    const half_t* Vb = VT + (((long)b * KVH_ + kvh) * HD_) * S_;

    const int mA    = lane & 15;
    const int khalf = (lane >> 4) * 8;
    const int nB    = lane & 15;
    const int kbB   = (lane >> 4) * 8;
    const int mbase = (lane >> 4) * 8;

    // ---- stage this wave's 16x128 Q tile via async DMA (wave-private) ----
    {
        const int qrow = lane & 15;
        const int qcol = (lane >> 4) * 64;            // 64 halves = 128B per lane
        const half_t* src = Qb + (long)(q0 + qrow) * HD_ + qcol;
        unsigned dst = lds_off(&Qs[wave][qrow * HD_ + qcol]);
        #pragma unroll
        for (int j = 0; j < 8; ++j)
            async_b128(dst + j * 16, src + j * 8);
        asm volatile("s_wait_asynccnt 0x0" ::: "memory");
    }
    // hoist all four Q A-fragments into registers
    AF aq[4];
    #pragma unroll
    for (int kk4 = 0; kk4 < 4; ++kk4) {
        const half_t* ql = &Qs[wave][mA * HD_ + kk4 * 32];
        aq[kk4].h[0] = *reinterpret_cast<const v8h*>(ql + khalf);
        aq[kk4].h[1] = *reinterpret_cast<const v8h*>(ql + 16 + khalf);
    }

    float m_run[8], l_run[8];
    v8f acc[8];
    #pragma unroll
    for (int r = 0; r < 8; ++r) { m_run[r] = -1e30f; l_run[r] = 0.0f; }
    #pragma unroll
    for (int t = 0; t < 8; ++t) acc[t] = (v8f){};

    for (int k0 = 0; k0 < q0 + 16; k0 += 32) {
        // ---- scores: S(16q x 32k) = Q(16 x 128) * K^T(128 x 32) ----
        v8f c0 = {}, c1 = {};
        #pragma unroll
        for (int kk4 = 0; kk4 < 4; ++kk4) {
            const int kk = kk4 * 32;
            AF b0, b1;
            const half_t* kr0 = Kb + (long)(k0 + nB) * HD_ + kk + kbB;
            b0.h[0] = *reinterpret_cast<const v8h*>(kr0);
            b0.h[1] = *reinterpret_cast<const v8h*>(kr0 + 16);
            const half_t* kr1 = Kb + (long)(k0 + 16 + nB) * HD_ + kk + kbB;
            b1.h[0] = *reinterpret_cast<const v8h*>(kr1);
            b1.h[1] = *reinterpret_cast<const v8h*>(kr1 + 16);

            c0 = __builtin_amdgcn_wmma_f32_16x16x32_f16(
                     false, aq[kk4].v, false, b0.v, (short)0, c0, false, false);
            c1 = __builtin_amdgcn_wmma_f32_16x16x32_f16(
                     false, aq[kk4].v, false, b1.v, (short)0, c1, false, false);
        }

        // ---- soft-cap, causal mask, online softmax ----
        float s0[8], s1[8], mx[8];
        #pragma unroll
        for (int r = 0; r < 8; ++r) {
            int qpos = q0 + mbase + r;
            int kp0  = k0 + nB;
            int kp1  = k0 + 16 + nB;
            float v0 = c0[r] * MULT_; v0 = MAXV_ * tanhf(v0 / MAXV_);
            float v1 = c1[r] * MULT_; v1 = MAXV_ * tanhf(v1 / MAXV_);
            if (kp0 > qpos) v0 = -10000.0f;
            if (kp1 > qpos) v1 = -10000.0f;
            s0[r] = v0; s1[r] = v1;
            mx[r] = fmaxf(v0, v1);
        }
        #pragma unroll
        for (int off = 1; off < 16; off <<= 1) {
            #pragma unroll
            for (int r = 0; r < 8; ++r)
                mx[r] = fmaxf(mx[r], __shfl_xor(mx[r], off, 32));
        }

        float ps[8];
        #pragma unroll
        for (int r = 0; r < 8; ++r) {
            float mnew  = fmaxf(m_run[r], mx[r]);
            float alpha = __expf(m_run[r] - mnew);
            m_run[r] = mnew;
            float p0 = __expf(s0[r] - mnew);
            float p1 = __expf(s1[r] - mnew);
            s0[r] = p0; s1[r] = p1;
            ps[r] = p0 + p1;
            l_run[r] *= alpha;
            #pragma unroll
            for (int t = 0; t < 8; ++t) acc[t][r] *= alpha;
        }
        #pragma unroll
        for (int off = 1; off < 16; off <<= 1) {
            #pragma unroll
            for (int r = 0; r < 8; ++r)
                ps[r] += __shfl_xor(ps[r], off, 32);
        }
        #pragma unroll
        for (int r = 0; r < 8; ++r) l_run[r] += ps[r];

        // ---- P: D-layout -> A-layout via per-wave LDS tile ----
        #pragma unroll
        for (int r = 0; r < 8; ++r) {
            Pt[wave][mbase + r][nB]      = (half_t)s0[r];
            Pt[wave][mbase + r][nB + 16] = (half_t)s1[r];
        }
        AF pa;
        pa.h[0] = *reinterpret_cast<const v8h*>(&Pt[wave][mA][khalf]);
        pa.h[1] = *reinterpret_cast<const v8h*>(&Pt[wave][mA][16 + khalf]);

        // ---- O(16 x 128) += P(16 x 32) * V(32 x 128), VT contiguous ----
        #pragma unroll
        for (int t = 0; t < 8; ++t) {
            AF bf;
            const half_t* vrow = Vb + (long)(t * 16 + nB) * S_ + k0 + kbB;
            bf.h[0] = *reinterpret_cast<const v8h*>(vrow);
            bf.h[1] = *reinterpret_cast<const v8h*>(vrow + 16);
            acc[t] = __builtin_amdgcn_wmma_f32_16x16x32_f16(
                         false, pa.v, false, bf.v, (short)0, acc[t], false, false);
        }
    }

    // ---- normalize + store to (B,S,H*HD) f16 ----
    #pragma unroll
    for (int t = 0; t < 8; ++t) {
        #pragma unroll
        for (int r = 0; r < 8; ++r) {
            int srow = q0 + mbase + r;
            O[(((long)b * S_ + srow) * H_ + h) * HD_ + t * 16 + nB] =
                (half_t)(acc[t][r] / l_run[r]);
        }
    }
}

// ------------------------------------------------------------------
// host-side launcher
// ------------------------------------------------------------------
extern "C" void kernel_launch(void* const* d_in, const int* in_sizes, int n_in,
                              void* d_out, int out_size, void* d_ws, size_t ws_size,
                              hipStream_t stream) {
    const float* hs = (const float*)d_in[0];
    // d_in[1] position_ids, d_in[2] attention_mask: recomputed on device
    const float* qw = (const float*)d_in[3];
    const float* kw = (const float*)d_in[4];
    const float* vw = (const float*)d_in[5];
    const float* ow = (const float*)d_in[6];
    float* out = (float*)d_out;

    char* wsp = (char*)d_ws;
    auto alloc = [&](size_t bytes) -> char* {
        char* p = wsp;
        wsp += (bytes + 255) & ~(size_t)255;
        return p;
    };

    half_t* hs16   = (half_t*)alloc((size_t)BS_ * D_  * 2);
    half_t* qw16   = (half_t*)alloc((size_t)D_  * QN_ * 2);
    half_t* kw16   = (half_t*)alloc((size_t)D_  * KN_ * 2);
    half_t* vw16   = (half_t*)alloc((size_t)D_  * KN_ * 2);
    half_t* ow16   = (half_t*)alloc((size_t)QN_ * D_  * 2);
    float*  qf     = (float*) alloc((size_t)BS_ * QN_ * 4);
    float*  kf     = (float*) alloc((size_t)BS_ * KN_ * 4);
    float*  vf     = (float*) alloc((size_t)BS_ * KN_ * 4);
    half_t* q16    = (half_t*)alloc((size_t)BS_ * QN_ * 2);
    half_t* k16    = (half_t*)alloc((size_t)BS_ * KN_ * 2);
    half_t* vT16   = (half_t*)alloc((size_t)BS_ * KN_ * 2);
    half_t* attn16 = (half_t*)alloc((size_t)BS_ * QN_ * 2);

    const int CT = 256, CG = 4096;
    cast_f32_f16<<<CG, CT, 0, stream>>>(hs, hs16, (long)BS_ * D_);
    cast_f32_f16<<<CG, CT, 0, stream>>>(qw, qw16, (long)D_ * QN_);
    cast_f32_f16<<<CG, CT, 0, stream>>>(kw, kw16, (long)D_ * KN_);
    cast_f32_f16<<<CG, CT, 0, stream>>>(vw, vw16, (long)D_ * KN_);
    cast_f32_f16<<<CG, CT, 0, stream>>>(ow, ow16, (long)QN_ * D_);

    // projections (block tile 128x128)
    gemm_f16_wmma<<<dim3(QN_ / BN_, BS_ / BM_), 256, 0, stream>>>(hs16, qw16, qf, BS_, QN_, D_);
    gemm_f16_wmma<<<dim3(KN_ / BN_, BS_ / BM_), 256, 0, stream>>>(hs16, kw16, kf, BS_, KN_, D_);
    gemm_f16_wmma<<<dim3(KN_ / BN_, BS_ / BM_), 256, 0, stream>>>(hs16, vw16, vf, BS_, KN_, D_);

    // rope + relayout to (B,heads,S,HD) f16; V transposed to (B,KVH,HD,S)
    rope_relayout   <<<CG, CT, 0, stream>>>(qf, q16, H_);
    rope_relayout   <<<CG, CT, 0, stream>>>(kf, k16, KVH_);
    cast_transpose_v<<<CG, CT, 0, stream>>>(vf, vT16);

    // attention: grid (S/128 q-tiles-of-8-waves, H, B)
    attn_wmma<<<dim3(S_ / 128, H_, B_), 256, 0, stream>>>(q16, k16, vT16, attn16);

    // output projection -> f32 d_out
    gemm_f16_wmma<<<dim3(D_ / BN_, BS_ / BM_), 256, 0, stream>>>(attn16, ow16, out, BS_, D_, QN_);
}
